// GCN_53266184405048
// MI455X (gfx1250) — matrix operand order repaired and checked by reading, not compiled, not run on hardware
//
#include <hip/hip_runtime.h>
#include <hip/hip_bf16.h>
#include <math.h>

// ---------------------------------------------------------------------------
// GCN layer for MI455X (gfx1250, wave32, WMMA).
//   out = sigmoid( relu( (D_in^-1/2 * A * D_out^-1/2 * X) @ Wgc + bgc ) @ Wfc + bfc )
// ---------------------------------------------------------------------------

#define N_NODES    50000
#define N_EDGES    800000
#define IN_FEATS   128
#define H_FEATS    256
#define NUM_CLS    16

#define WAVES_PER_BLOCK 4
#define H_PAD 264   // 256 halves + 8 pad (keeps 16B alignment, reduces bank conflicts)

typedef __attribute__((ext_vector_type(16))) _Float16 v16h;
typedef __attribute__((ext_vector_type(8)))  _Float16 v8h;
typedef __attribute__((ext_vector_type(8)))  float    v8f;

// ---------------------------------------------------------------------------
// 1) zero agg accumulator + degree buffers
// ---------------------------------------------------------------------------
__global__ void gcn_zero(float4* __restrict__ agg4, int n4,
                         float* __restrict__ dout, float* __restrict__ din, int n) {
  int t = blockIdx.x * blockDim.x + threadIdx.x;
  if (t < n4) agg4[t] = make_float4(0.f, 0.f, 0.f, 0.f);
  if (t < n)  { dout[t] = 0.f; din[t] = 0.f; }
}

// ---------------------------------------------------------------------------
// 2) degree counts (float atomics; counts < 2^24 so exact)
// ---------------------------------------------------------------------------
__global__ void gcn_degrees(const int* __restrict__ src, const int* __restrict__ dst,
                            float* __restrict__ dout, float* __restrict__ din) {
  int e = blockIdx.x * blockDim.x + threadIdx.x;
  if (e >= N_EDGES) return;
  atomicAdd(&dout[src[e]], 1.0f);
  atomicAdd(&din[dst[e]], 1.0f);
}

// ---------------------------------------------------------------------------
// 3) deg -> rsqrt(max(deg,1)) in place
// ---------------------------------------------------------------------------
__global__ void gcn_rinv(float* __restrict__ dout, float* __restrict__ din) {
  int t = blockIdx.x * blockDim.x + threadIdx.x;
  if (t >= N_NODES) return;
  dout[t] = rsqrtf(fmaxf(dout[t], 1.0f));
  din[t]  = rsqrtf(fmaxf(din[t],  1.0f));
}

// ---------------------------------------------------------------------------
// 4) pack weights into wave32 WMMA B-fragment layout, f32 -> f16.
//    B (KxN) fragment layout for v_wmma_f32_16x16x32_f16:
//      lane L: column n = (L&15);  element e: 0..7 -> K = 8*(L>=16)+e, 8..15 -> K = 8*(L>=16)+16+(e-8)
//    pWgc index = ((kt*16 + nt)*32 + L)*16 + e     (kt: K/32, nt: N/16)
//    pWfc index = ((kt)*32 + L)*16 + e             (single 16-wide N tile)
// ---------------------------------------------------------------------------
__global__ void gcn_pack_w(const float* __restrict__ Wgc, const float* __restrict__ Wfc,
                           _Float16* __restrict__ pgc, _Float16* __restrict__ pfc) {
  int t = blockIdx.x * blockDim.x + threadIdx.x;
  const int NGC = (IN_FEATS / 32) * (H_FEATS / 16) * 32 * 16;  // 32768
  const int NFC = (H_FEATS / 32) * 32 * 16;                    // 4096
  if (t < NGC) {
    int e  = t & 15;
    int L  = (t >> 4) & 31;
    int nt = (t >> 9) & 15;
    int kt = t >> 13;
    int k = kt * 32 + ((L >> 4) << 3) + (e < 8 ? e : e + 8);
    int n = nt * 16 + (L & 15);
    pgc[t] = (_Float16)Wgc[k * H_FEATS + n];
  } else if (t < NGC + NFC) {
    int u  = t - NGC;
    int e  = u & 15;
    int L  = (u >> 4) & 31;
    int kt = u >> 9;
    int k = kt * 32 + ((L >> 4) << 3) + (e < 8 ? e : e + 8);
    int n = L & 15;
    pfc[u] = (_Float16)Wfc[k * NUM_CLS + n];
  }
}

// ---------------------------------------------------------------------------
// 5) edge aggregation: agg[dst] += in_feat[src] * rsqrt(deg_out[src])
//    one wave per edge, lane covers 4 features (float4 gather, 4 f32 atomics).
//    Both in_feat (25.6MB) and agg (25.6MB) are L2-resident on MI455X (192MB L2).
// ---------------------------------------------------------------------------
__global__ void gcn_edge_agg(const float* __restrict__ x,
                             const int* __restrict__ src, const int* __restrict__ dst,
                             const float* __restrict__ rout,
                             float* __restrict__ agg) {
  int warp = (blockIdx.x * blockDim.x + threadIdx.x) >> 5;
  int lane = threadIdx.x & 31;
  if (warp >= N_EDGES) return;
  int s = src[warp];
  int d = dst[warp];
  float sc = rout[s];
  float4 v = *(const float4*)(x + s * IN_FEATS + lane * 4);
  float* o = agg + d * IN_FEATS + lane * 4;
  atomicAdd(o + 0, v.x * sc);
  atomicAdd(o + 1, v.y * sc);
  atomicAdd(o + 2, v.z * sc);
  atomicAdd(o + 3, v.w * sc);
}

// ---------------------------------------------------------------------------
// 6) fused dual-GEMM + activations, one 16-node tile per wave.
//    GEMM1: h = relu(agg_scaled @ Wgc + bgc)   [16x128]@[128x256]
//    GEMM2: out = sigmoid(h @ Wfc + bfc)       [16x256]@[256x16]
//    A fragment (16x32 f16, MxK):
//      lane L: row m = L&15; elements e: 0..7 -> K = 8*(L>=16)+e,
//                                        8..15 -> K = 8*(L>=16)+16+(e-8)
//    C/D: VGPR r -> row r + 8*(L>=16), col = L&15.
// ---------------------------------------------------------------------------
__global__ __launch_bounds__(WAVES_PER_BLOCK * 32)
void gcn_wmma(const float* __restrict__ agg,
              const float* __restrict__ rin_in,
              const float* __restrict__ bgc,
              const float* __restrict__ bfc,
              const _Float16* __restrict__ pgc,
              const _Float16* __restrict__ pfc,
              float* __restrict__ out) {
  __shared__ __align__(16) _Float16 hbuf[WAVES_PER_BLOCK][16][H_PAD];

  int lane = threadIdx.x & 31;
  int wv   = threadIdx.x >> 5;
  int tile = blockIdx.x * WAVES_PER_BLOCK + wv;
  int row0 = tile * 16;
  if (row0 >= N_NODES) return;

  int m  = lane & 15;     // A-row / D-column index for this lane
  int hi = lane >> 4;     // which half of the wave
  int row = row0 + m;
  float rin = rin_in[row];

  // ---- A fragments for GEMM1: agg row * rsqrt(deg_in), f32 -> f16
  v16h a[4];
  const float* arow = agg + row * IN_FEATS;
#pragma unroll
  for (int kt = 0; kt < 4; ++kt) {
    int bk = kt * 32 + hi * 8;
    float4 f0 = *(const float4*)(arow + bk);
    float4 f1 = *(const float4*)(arow + bk + 4);
    float4 g0 = *(const float4*)(arow + bk + 16);
    float4 g1 = *(const float4*)(arow + bk + 20);
    float tf[16] = {f0.x, f0.y, f0.z, f0.w, f1.x, f1.y, f1.z, f1.w,
                    g0.x, g0.y, g0.z, g0.w, g1.x, g1.y, g1.z, g1.w};
#pragma unroll
    for (int e = 0; e < 16; ++e) a[kt][e] = (_Float16)(tf[e] * rin);
  }

  // ---- GEMM1: 16 N-tiles x 4 K-tiles = 64 WMMAs; bias+relu -> LDS (f16)
  const v16h* wgc = (const v16h*)pgc;
#pragma unroll
  for (int nt = 0; nt < 16; ++nt) {
    v8f acc = {};
#pragma unroll
    for (int kt = 0; kt < 4; ++kt) {
      v16h b = wgc[(kt * 16 + nt) * 32 + lane];
      acc = __builtin_amdgcn_wmma_f32_16x16x32_f16(
          false, a[kt], false, b, (short)0, acc, false, false);
    }
    float bias = bgc[nt * 16 + m];
    int rb = hi * 8;
#pragma unroll
    for (int r = 0; r < 8; ++r) {
      float v = fmaxf(acc[r] + bias, 0.0f);
      hbuf[wv][rb + r][nt * 16 + m] = (_Float16)v;
    }
  }

  // Same-wave LDS RAW: make sure all DS stores have landed before reload.
  asm volatile("s_wait_dscnt 0" ::: "memory");

  // ---- GEMM2: 8 K-tiles over H=256, single 16-wide N tile (classes)
  const v16h* wfc = (const v16h*)pfc;
  v8f acc2 = {};
#pragma unroll
  for (int kt = 0; kt < 8; ++kt) {
    int bk = kt * 32 + hi * 8;
    v8h lo = *(const v8h*)&hbuf[wv][m][bk];
    v8h hp = *(const v8h*)&hbuf[wv][m][bk + 16];
    v16h a2;
#pragma unroll
    for (int e = 0; e < 8; ++e) { a2[e] = lo[e]; a2[8 + e] = hp[e]; }
    v16h b2 = wfc[kt * 32 + lane];
    acc2 = __builtin_amdgcn_wmma_f32_16x16x32_f16(
        false, a2, false, b2, (short)0, acc2, false, false);
  }

  // ---- epilogue: bias + sigmoid, coalesced f32 stores
  float bias2 = bfc[m];
  int rb = hi * 8;
#pragma unroll
  for (int r = 0; r < 8; ++r) {
    float v = acc2[r] + bias2;
    float s = 1.0f / (1.0f + __expf(-v));
    out[(row0 + rb + r) * NUM_CLS + m] = s;
  }
}

// ---------------------------------------------------------------------------
// launch
// ---------------------------------------------------------------------------
extern "C" void kernel_launch(void* const* d_in, const int* in_sizes, int n_in,
                              void* d_out, int out_size, void* d_ws, size_t ws_size,
                              hipStream_t stream) {
  const float* in_feat = (const float*)d_in[0];
  const int*   src     = (const int*)d_in[1];
  const int*   dst     = (const int*)d_in[2];
  const float* W_gc    = (const float*)d_in[3];
  const float* b_gc    = (const float*)d_in[4];
  const float* W_fc    = (const float*)d_in[5];
  const float* b_fc    = (const float*)d_in[6];
  float* out = (float*)d_out;

  // workspace layout (all offsets 32B-aligned)
  char* ws = (char*)d_ws;
  float*    agg  = (float*)(ws);                         // 50000*128*4 = 25,600,000 B
  float*    dout = (float*)(ws + 25600000);              //   200,000 B
  float*    din  = (float*)(ws + 25800000);              //   200,000 B
  _Float16* pgc  = (_Float16*)(ws + 26000000);           //    65,536 B
  _Float16* pfc  = (_Float16*)(ws + 26065536);           //     8,192 B

  const int n4 = N_NODES * IN_FEATS / 4;  // 1,600,000 float4 zeros

  // 1) zero accumulators
  gcn_zero<<<(n4 + 255) / 256, 256, 0, stream>>>((float4*)agg, n4, dout, din, N_NODES);
  // 2) degrees
  gcn_degrees<<<(N_EDGES + 255) / 256, 256, 0, stream>>>(src, dst, dout, din);
  // 3) rsqrt(clamp(deg,1))
  gcn_rinv<<<(N_NODES + 255) / 256, 256, 0, stream>>>(dout, din);
  // 4) pack weights into WMMA fragment layout (36,864 elements total)
  gcn_pack_w<<<(36864 + 255) / 256, 256, 0, stream>>>(W_gc, W_fc, pgc, pfc);
  // 5) edge aggregation: one wave per edge
  gcn_edge_agg<<<(N_EDGES * 32) / 256, 256, 0, stream>>>(in_feat, src, dst, dout, agg);
  // 6) fused WMMA dual-GEMM + activations: one 16-node tile per wave
  const int tiles  = (N_NODES + 15) / 16;                        // 3125
  const int blocks = (tiles + WAVES_PER_BLOCK - 1) / WAVES_PER_BLOCK;  // 782
  gcn_wmma<<<blocks, WAVES_PER_BLOCK * 32, 0, stream>>>(agg, din, b_gc, b_fc,
                                                        pgc, pfc, out);
}
